// GraphConvolutionLayer_52982716564118
// MI455X (gfx1250) — compile-verified, matrix operand order checked by reading, not run on hardware
//
#include <hip/hip_runtime.h>

#define NN 8192     // nodes
#define F  256      // features (in == out)

typedef _Float16 h4   __attribute__((ext_vector_type(4)));
typedef _Float16 h8   __attribute__((ext_vector_type(8)));
typedef _Float16 v16h __attribute__((ext_vector_type(16)));
typedef float    v8f  __attribute__((ext_vector_type(8)));
typedef float    f32x4 __attribute__((ext_vector_type(4)));

#define ADJ_STRIDE 40    // 32 K halves + pad, 80B rows (16B aligned)
#define XT_STRIDE  40    // 32 K halves + pad
#define S_STRIDE   264   // 256 halves + pad, 528B rows (16B aligned)

// Build a 16-half WMMA fragment from two contiguous 16B runs:
// per ISA layout, lane's halves [0..7] = K at (8*halfSel), halves [8..15] = +16.
static __device__ inline v16h load_frag(const _Float16* __restrict__ rowBase, int halfSel) {
    h8 lo = *(const h8*)(rowBase + halfSel * 8);
    h8 hi = *(const h8*)(rowBase + halfSel * 8 + 16);
    return __builtin_shufflevector(lo, hi, 0,1,2,3,4,5,6,7,8,9,10,11,12,13,14,15);
}

// Pre-transpose + convert kernel weights: WkT[n][k] = (f16) W[k][n]
__global__ void gcn_prep_wkT(const float* __restrict__ kern, _Float16* __restrict__ wkT) {
    int idx = blockIdx.x * blockDim.x + threadIdx.x;   // 0 .. 65535
    int n = idx >> 8, k = idx & 255;
    wkT[idx] = (_Float16)kern[k * F + n];
}

// One block = 64 output rows x all 256 columns. 512 threads = 16 waves.
// Fuses: rowsum(adj), agg = adj@x + beta*x, scale, and aggS @ W + bias.
__global__ __launch_bounds__(512)
void gcn_fused(const float* __restrict__ x, const float* __restrict__ adj,
               const float* __restrict__ bias, const float* __restrict__ beta,
               const _Float16* __restrict__ wkT, float* __restrict__ out) {
    __shared__ _Float16 adjL[64 * ADJ_STRIDE];   // 5 KB
    __shared__ _Float16 xTL[F * XT_STRIDE];      // 20 KB, transposed [col][k]
    __shared__ _Float16 S[64 * S_STRIDE];        // 33 KB, scaled agg (f16)
    __shared__ float scaleL[64];
    __shared__ float betaL[64];

    const int tid  = threadIdx.x;
    const int lane = tid & 31;
    const int wave = tid >> 5;
    const int row0 = blockIdx.x * 64;

    if (tid < 64) betaL[tid] = beta[row0 + tid];

    // adj-tile loader mapping: each thread owns one (row, 4-wide K slot) for ALL k0
    const int aRow = tid >> 3;            // 0..63
    const int aK4  = (tid & 7) * 4;       // 0..28
    // x-tile loader mapping: 4 cols x 4 k's per thread
    const int xC4 = tid & 63;             // column group of 4
    const int xQ  = tid >> 6;             // k group of 4 (0..7)

    // wave tiling: row tile (16 rows) x 4 column tiles (64 cols)
    const int rt      = wave & 3;
    const int cg      = wave >> 2;
    const int halfSel = lane >> 4;
    const int lm      = lane & 15;

    float rowsum = 0.f;
    v8f acc[4];
#pragma unroll
    for (int j = 0; j < 4; ++j) acc[j] = (v8f){0,0,0,0,0,0,0,0};

    // ---------------- K loop over adj columns: agg = adj @ x ----------------
    for (int k0 = 0; k0 < NN; k0 += 32) {
        __syncthreads();  // previous iter's fragment reads done (also covers betaL)

        // adj tile 64x32: streaming NT load, convert f32->f16, fused row-sum
        f32x4 av = __builtin_nontemporal_load(
            (const f32x4*)(adj + (size_t)(row0 + aRow) * NN + k0 + aK4));
        rowsum += av[0] + av[1] + av[2] + av[3];
        h4 apk = {(_Float16)av[0], (_Float16)av[1], (_Float16)av[2], (_Float16)av[3]};
        *(h4*)(adjL + aRow * ADJ_STRIDE + aK4) = apk;

        // x tile 32x256 -> LDS transposed [col][k]; x stays L2-resident (RT loads)
        f32x4 xv[4];
#pragma unroll
        for (int jj = 0; jj < 4; ++jj)
            xv[jj] = *(const f32x4*)(x + (size_t)(k0 + xQ * 4 + jj) * F + xC4 * 4);
#pragma unroll
        for (int m = 0; m < 4; ++m) {
            h4 pk = {(_Float16)xv[0][m], (_Float16)xv[1][m],
                     (_Float16)xv[2][m], (_Float16)xv[3][m]};
            *(h4*)(xTL + (xC4 * 4 + m) * XT_STRIDE + xQ * 4) = pk;
        }

        __syncthreads();

        v16h a = load_frag(adjL + (rt * 16 + lm) * ADJ_STRIDE, halfSel);
#pragma unroll
        for (int j = 0; j < 4; ++j) {
            v16h b = load_frag(xTL + (cg * 64 + j * 16 + lm) * XT_STRIDE, halfSel);
            acc[j] = __builtin_amdgcn_wmma_f32_16x16x32_f16(
                false, a, false, b, (short)0, acc[j], false, false);
        }
    }

    // ---------------- row-sum reduction -> scale = 1/(rowsum + beta) --------
    rowsum += __shfl_xor(rowsum, 1);
    rowsum += __shfl_xor(rowsum, 2);
    rowsum += __shfl_xor(rowsum, 4);
    if ((tid & 7) == 0) scaleL[aRow] = 1.0f / (rowsum + betaL[aRow]);
    __syncthreads();

    // ---------------- epilogue A: S = scale * (acc + beta * x) (f16) --------
#pragma unroll
    for (int j = 0; j < 4; ++j) {
        int col = cg * 64 + j * 16 + lm;
#pragma unroll
        for (int v = 0; v < 8; ++v) {
            int rl = rt * 16 + v + 8 * halfSel;               // C-layout: VGPR v -> M = v (+8)
            float val = acc[j][v] + betaL[rl] * x[(size_t)(row0 + rl) * F + col];
            S[rl * S_STRIDE + col] = (_Float16)(scaleL[rl] * val);
        }
    }
    __syncthreads();

    // ---------------- stage B: out = S @ W + bias ---------------------------
    v8f acc2[4];
#pragma unroll
    for (int j = 0; j < 4; ++j) acc2[j] = (v8f){0,0,0,0,0,0,0,0};

    for (int k0 = 0; k0 < F; k0 += 32) {
        v16h a2 = load_frag(S + (rt * 16 + lm) * S_STRIDE + k0, halfSel);
#pragma unroll
        for (int j = 0; j < 4; ++j) {
            v16h b2 = load_frag(wkT + (size_t)(cg * 64 + j * 16 + lm) * F + k0, halfSel);
            acc2[j] = __builtin_amdgcn_wmma_f32_16x16x32_f16(
                false, a2, false, b2, (short)0, acc2[j], false, false);
        }
    }

#pragma unroll
    for (int j = 0; j < 4; ++j) {
        int col = cg * 64 + j * 16 + lm;
        float bc = bias[col];
#pragma unroll
        for (int v = 0; v < 8; ++v) {
            int rl = rt * 16 + v + 8 * halfSel;
            out[(size_t)(row0 + rl) * F + col] = acc2[j][v] + bc;
        }
    }
}

extern "C" void kernel_launch(void* const* d_in, const int* in_sizes, int n_in,
                              void* d_out, int out_size, void* d_ws, size_t ws_size,
                              hipStream_t stream) {
    (void)in_sizes; (void)n_in; (void)out_size; (void)ws_size;
    const float* x    = (const float*)d_in[0];   // [8192, 256]
    const float* adj  = (const float*)d_in[1];   // [8192, 8192]
    const float* kern = (const float*)d_in[2];   // [256, 256]
    const float* bias = (const float*)d_in[3];   // [256]
    const float* beta = (const float*)d_in[4];   // [8192]
    float* out = (float*)d_out;                  // [8192, 256]

    _Float16* wkT = (_Float16*)d_ws;             // 256*256 f16 = 128 KB

    gcn_prep_wkT<<<(F * F) / 256, 256, 0, stream>>>(kern, wkT);
    gcn_fused<<<NN / 64, 512, 0, stream>>>(x, adj, bias, beta, wkT, out);
}
